// EViT_72748156060124
// MI455X (gfx1250) — matrix operand order repaired and checked by reading, not compiled, not run on hardware
//
#include <hip/hip_runtime.h>
#include <math.h>

typedef __attribute__((ext_vector_type(16))) _Float16 v16h;
typedef __attribute__((ext_vector_type(8)))  _Float16 v8h;
typedef __attribute__((ext_vector_type(8)))  float    v8f;

// ---------------- problem constants ----------------
#define BB      32
#define CC      3
#define IMG     224
#define PP      16
#define GRID_   14
#define NPATCH  196
#define DD      768
#define HH      12
#define NLAY    12
#define DFF     3072
#define DHD     64
#define NCLS    100
#define SELL    3
#define KEEP    137
#define L0      197      // tokens before prune
#define L1      138      // tokens after prune (1 + KEEP)
#define LP0     200      // padded score stride (ceil4)
#define LP1     140

__device__ __forceinline__ float gelu_f(float x) {
    float x3 = x * x * x;
    return 0.5f * x * (1.0f + tanhf(0.7978845608028654f * (x + 0.044715f * x3)));
}

// =====================================================================
// Generic strided-batched WMMA GEMM:
//   C[z] = epilogue( alpha * A[z] @ op(B[z]) )
//   epilogue: (+bias[n]) -> (gelu?) -> (+res[z][m][n])
// Block tile 128x64, K-step 32, 8 waves (wave32) in a 4(M) x 2(N) grid;
// each wave computes a 32x32 tile = 4 v_wmma_f32_16x16x32_f16 per K-step
// from 2 A-frags + 2 B-frags (8 ds_load_b128 -> 2 LDS loads per WMMA).
// Interior tiles: unconditional float4 global loads + register double
// buffering; B staged transposed in LDS so fragments read contiguous
// 16B chunks.
// =====================================================================
#define TM 128
#define TN 64
#define TK 32
#define LDSP (TK + 8)    // half pitch: 40 halves = 80B (16B aligned)

__global__ __launch_bounds__(256)
void wmma_gemm_kernel(const float* __restrict__ A, int lda, long sAb, long sAh,
                      const float* __restrict__ Bm, int ldb, long sBb, long sBh, int transB,
                      float* C, int ldc, long sCb, long sCh,
                      const float* __restrict__ bias,
                      const float* res, int ldr, long sRb, long sRh,
                      int M, int N, int K, int Hs, float alpha, int act)
{
    const int z  = blockIdx.z;
    const int zb = z / Hs, zh = z % Hs;
    A  += (long)zb * sAb + (long)zh * sAh;
    Bm += (long)zb * sBb + (long)zh * sBh;
    C  += (long)zb * sCb + (long)zh * sCh;
    if (res) res += (long)zb * sRb + (long)zh * sRh;

    __shared__ __align__(16) _Float16 As [TM][LDSP];   // As[m][k]
    __shared__ __align__(16) _Float16 BsT[TN][LDSP];   // BsT[n][k]

    const int tid  = threadIdx.x;
    const int lane = tid & 31;
    const int wid  = tid >> 5;
    const int wm   = wid & 3;        // wave row block (32 rows)
    const int wn   = wid >> 2;       // wave col block (32 cols)
    const int m0   = blockIdx.y * TM;
    const int n0   = blockIdx.x * TN;

    const bool fullM = (m0 + TM <= M);
    const bool fullN = (n0 + TN <= N);
    const bool vecA  = ((lda & 3) == 0);
    const bool vecB  = ((ldb & 3) == 0);

    // staging coordinates: each thread owns 4 quads of A and 2 quads of B
    const int rowA = tid >> 3;            // 0..31 (+32*i)
    const int kqA  = (tid & 7) * 4;       // 0,4,..,28
    const int nnB  = tid >> 3;            // 0..31 (+32*i)
    const int kqB  = (tid & 7) * 4;

    float4 ra[4], rb[2];

    auto fetchA = [&](int k0) {
        const bool fullK = (k0 + TK <= K);
        #pragma unroll
        for (int i = 0; i < 4; ++i) {
            const int gm = m0 + rowA + 32 * i;
            if (fullM && fullK && vecA) {
                ra[i] = *(const float4*)(A + (long)gm * lda + (k0 + kqA));
            } else {
                const int gk = k0 + kqA;
                float4 v;
                v.x = (gm < M && gk + 0 < K) ? A[(long)gm * lda + gk + 0] : 0.0f;
                v.y = (gm < M && gk + 1 < K) ? A[(long)gm * lda + gk + 1] : 0.0f;
                v.z = (gm < M && gk + 2 < K) ? A[(long)gm * lda + gk + 2] : 0.0f;
                v.w = (gm < M && gk + 3 < K) ? A[(long)gm * lda + gk + 3] : 0.0f;
                ra[i] = v;
            }
        }
    };
    auto fetchB = [&](int k0) {
        const bool fullK = (k0 + TK <= K);
        #pragma unroll
        for (int i = 0; i < 2; ++i) {
            const int gn = n0 + nnB + 32 * i;
            const int gk = k0 + kqB;
            if (transB) {
                if (fullN && fullK && vecB) {
                    rb[i] = *(const float4*)(Bm + (long)gn * ldb + gk);
                } else {
                    float4 v;
                    v.x = (gn < N && gk + 0 < K) ? Bm[(long)gn * ldb + gk + 0] : 0.0f;
                    v.y = (gn < N && gk + 1 < K) ? Bm[(long)gn * ldb + gk + 1] : 0.0f;
                    v.z = (gn < N && gk + 2 < K) ? Bm[(long)gn * ldb + gk + 2] : 0.0f;
                    v.w = (gn < N && gk + 3 < K) ? Bm[(long)gn * ldb + gk + 3] : 0.0f;
                    rb[i] = v;
                }
            } else {
                // strided in k, coalesced across lanes in n
                float4 v;
                if (fullN && fullK) {
                    v.x = Bm[(long)(gk + 0) * ldb + gn];
                    v.y = Bm[(long)(gk + 1) * ldb + gn];
                    v.z = Bm[(long)(gk + 2) * ldb + gn];
                    v.w = Bm[(long)(gk + 3) * ldb + gn];
                } else {
                    v.x = (gn < N && gk + 0 < K) ? Bm[(long)(gk + 0) * ldb + gn] : 0.0f;
                    v.y = (gn < N && gk + 1 < K) ? Bm[(long)(gk + 1) * ldb + gn] : 0.0f;
                    v.z = (gn < N && gk + 2 < K) ? Bm[(long)(gk + 2) * ldb + gn] : 0.0f;
                    v.w = (gn < N && gk + 3 < K) ? Bm[(long)(gk + 3) * ldb + gn] : 0.0f;
                }
                rb[i] = v;
            }
        }
    };
    auto stage = [&]() {
        #pragma unroll
        for (int i = 0; i < 4; ++i) {
            _Float16* pa = &As[rowA + 32 * i][kqA];
            pa[0] = (_Float16)ra[i].x; pa[1] = (_Float16)ra[i].y;
            pa[2] = (_Float16)ra[i].z; pa[3] = (_Float16)ra[i].w;
        }
        #pragma unroll
        for (int i = 0; i < 2; ++i) {
            _Float16* pb = &BsT[nnB + 32 * i][kqB];
            pb[0] = (_Float16)rb[i].x; pb[1] = (_Float16)rb[i].y;
            pb[2] = (_Float16)rb[i].z; pb[3] = (_Float16)rb[i].w;
        }
    };

    // A fragment loader: row-contiguous 16B chunks from As
    auto loadAfrag = [&](int rbase) -> v16h {
        const int row = rbase + (lane & 15);
        const int kb  = (lane >> 4) * 8;
        v8h lo = *(const v8h*)&As[row][kb];
        v8h hi = *(const v8h*)&As[row][kb + 16];
        v16h f;
        #pragma unroll
        for (int e = 0; e < 8; ++e) { f[e] = lo[e]; f[8 + e] = hi[e]; }
        return f;
    };
    // B fragment loader: element e <-> K = (lane/16)*8 + e (e<8), 16 + (lane/16)*8 + (e-8) (e>=8)
    auto loadBfrag = [&](int cbase) -> v16h {
        const int col = cbase + (lane & 15);
        const int kb  = (lane >> 4) * 8;
        v8h lo = *(const v8h*)&BsT[col][kb];
        v8h hi = *(const v8h*)&BsT[col][kb + 16];
        v16h f;
        #pragma unroll
        for (int e = 0; e < 8; ++e) { f[e] = lo[e]; f[8 + e] = hi[e]; }
        return f;
    };

    v8f c00 = {}, c01 = {}, c10 = {}, c11 = {};

    fetchA(0);
    fetchB(0);

    for (int k0 = 0; k0 < K; k0 += TK) {
        stage();
        __syncthreads();

        // prefetch next tile into registers (overlaps with WMMA below)
        if (k0 + TK < K) { fetchA(k0 + TK); fetchB(k0 + TK); }

        v16h af0 = loadAfrag(wm * 32);
        v16h af1 = loadAfrag(wm * 32 + 16);
        v16h bf0 = loadBfrag(wn * 32);
        v16h bf1 = loadBfrag(wn * 32 + 16);

        c00 = __builtin_amdgcn_wmma_f32_16x16x32_f16(false, af0, false, bf0,
                                                     (short)0, c00, false, false);
        c01 = __builtin_amdgcn_wmma_f32_16x16x32_f16(false, af0, false, bf1,
                                                     (short)0, c01, false, false);
        c10 = __builtin_amdgcn_wmma_f32_16x16x32_f16(false, af1, false, bf0,
                                                     (short)0, c10, false, false);
        c11 = __builtin_amdgcn_wmma_f32_16x16x32_f16(false, af1, false, bf1,
                                                     (short)0, c11, false, false);
        __syncthreads();
    }

    // epilogue; C layout: vgpr r -> M = r + 8*(lane>=16), N = lane%16
    const int rowb = m0 + wm * 32 + ((lane >> 4) * 8);
    const int colb = n0 + wn * 32 + (lane & 15);
    #pragma unroll
    for (int half = 0; half < 2; ++half) {
        const v8f* cl = half ? &c10 : &c00;
        const v8f* ch = half ? &c11 : &c01;
        #pragma unroll
        for (int r = 0; r < 8; ++r) {
            int gm = rowb + 16 * half + r;
            if (gm >= M) continue;
            int gn = colb;
            if (gn < N) {
                float t = alpha * (*cl)[r];
                if (bias) t += bias[gn];
                if (act)  t = gelu_f(t);
                if (res)  t += res[(long)gm * ldr + gn];
                C[(long)gm * ldc + gn] = t;
            }
            gn = colb + 16;
            if (gn < N) {
                float t = alpha * (*ch)[r];
                if (bias) t += bias[gn];
                if (act)  t = gelu_f(t);
                if (res)  t += res[(long)gm * ldr + gn];
                C[(long)gm * ldc + gn] = t;
            }
        }
    }
}

// ---------------- LayerNorm (one block per token row) ----------------
__global__ __launch_bounds__(256)
void ln_kernel(const float* __restrict__ x, const float* __restrict__ g,
               const float* __restrict__ b, float* __restrict__ y, int Dp)
{
    long row = blockIdx.x;
    const float* xr = x + row * (long)Dp;
    float* yr = y + row * (long)Dp;
    __shared__ float s1[256], s2[256];
    float a = 0.f, c = 0.f;
    for (int i = threadIdx.x; i < Dp; i += 256) { float v = xr[i]; a += v; c += v * v; }
    s1[threadIdx.x] = a; s2[threadIdx.x] = c; __syncthreads();
    for (int st = 128; st > 0; st >>= 1) {
        if (threadIdx.x < st) { s1[threadIdx.x] += s1[threadIdx.x + st];
                                s2[threadIdx.x] += s2[threadIdx.x + st]; }
        __syncthreads();
    }
    float mean = s1[0] / Dp;
    float var  = s2[0] / Dp - mean * mean;
    float rs   = rsqrtf(var + 1e-6f);
    for (int i = threadIdx.x; i < Dp; i += 256)
        yr[i] = (xr[i] - mean) * rs * g[i] + b[i];
}

// --------- row softmax; rows have stride ldp, valid length L ---------
__global__ __launch_bounds__(256)
void softmax_kernel(float* __restrict__ s, int L, int ldp)
{
    long row = blockIdx.x;
    float* p = s + row * (long)ldp;
    __shared__ float red[256];
    float m = -1e30f;
    for (int i = threadIdx.x; i < L; i += 256) m = fmaxf(m, p[i]);
    red[threadIdx.x] = m; __syncthreads();
    for (int st = 128; st > 0; st >>= 1) {
        if (threadIdx.x < st) red[threadIdx.x] = fmaxf(red[threadIdx.x], red[threadIdx.x + st]);
        __syncthreads();
    }
    m = red[0]; __syncthreads();
    float sum = 0.f;
    for (int i = threadIdx.x; i < L; i += 256) { float e = expf(p[i] - m); p[i] = e; sum += e; }
    red[threadIdx.x] = sum; __syncthreads();
    for (int st = 128; st > 0; st >>= 1) {
        if (threadIdx.x < st) red[threadIdx.x] += red[threadIdx.x + st];
        __syncthreads();
    }
    float inv = 1.0f / red[0];
    for (int i = threadIdx.x; i < L; i += 256) p[i] *= inv;
}

// ---------------- im2col: (B,C,224,224) -> (B*196, 768) -------------
__global__ __launch_bounds__(256)
void im2col_kernel(const float* __restrict__ in, float* __restrict__ out)
{
    long i = (long)blockIdx.x * 256 + threadIdx.x;
    const long total = (long)BB * NPATCH * DD;
    if (i >= total) return;
    int d = (int)(i % DD);
    long t = i / DD;
    int p = (int)(t % NPATCH);
    int b = (int)(t / NPATCH);
    int c  = d / (PP * PP);
    int rr = d % (PP * PP);
    int py = rr / PP, px = rr % PP;
    int gy = p / GRID_, gx = p % GRID_;
    out[i] = in[(((long)b * CC + c) * IMG + (gy * PP + py)) * IMG + (gx * PP + px)];
}

// ------------- assemble x = concat(cls, patches) + pos_embed --------
__global__ __launch_bounds__(256)
void build_x_kernel(const float* __restrict__ patch_out, const float* __restrict__ cls,
                    const float* __restrict__ pos, float* __restrict__ x)
{
    int b = blockIdx.x / L0;
    int l = blockIdx.x % L0;
    float* xr = x + ((long)b * L0 + l) * DD;
    const float* pr = pos + (long)l * DD;
    if (l == 0) {
        for (int d = threadIdx.x; d < DD; d += 256) xr[d] = cls[d] + pr[d];
    } else {
        const float* po = patch_out + ((long)b * NPATCH + (l - 1)) * DD;
        for (int d = threadIdx.x; d < DD; d += 256) xr[d] = po[d] + pr[d];
    }
}

// ---------------- ||cls_b|| ------------------------------------------
__global__ __launch_bounds__(256)
void clsnorm_kernel(const float* __restrict__ x, float* __restrict__ out, int L)
{
    int b = blockIdx.x;
    const float* c = x + (long)b * L * DD;
    __shared__ float red[256];
    float s = 0.f;
    for (int i = threadIdx.x; i < DD; i += 256) { float v = c[i]; s += v * v; }
    red[threadIdx.x] = s; __syncthreads();
    for (int st = 128; st > 0; st >>= 1) {
        if (threadIdx.x < st) red[threadIdx.x] += red[threadIdx.x + st];
        __syncthreads();
    }
    if (threadIdx.x == 0) out[b] = sqrtf(red[0]);
}

// ---------------- cosine similarity cls vs patch ---------------------
__global__ __launch_bounds__(256)
void sim_kernel(const float* __restrict__ x, const float* __restrict__ clsn,
                float* __restrict__ sim, int L)
{
    int b = blockIdx.x / (L - 1);
    int p = blockIdx.x % (L - 1);
    const float* c = x + (long)b * L * DD;
    const float* t = c + (long)(1 + p) * DD;
    __shared__ float r1[256], r2[256];
    float dot = 0.f, ss = 0.f;
    for (int i = threadIdx.x; i < DD; i += 256) { float tv = t[i]; dot += c[i] * tv; ss += tv * tv; }
    r1[threadIdx.x] = dot; r2[threadIdx.x] = ss; __syncthreads();
    for (int st = 128; st > 0; st >>= 1) {
        if (threadIdx.x < st) { r1[threadIdx.x] += r1[threadIdx.x + st];
                                r2[threadIdx.x] += r2[threadIdx.x + st]; }
        __syncthreads();
    }
    if (threadIdx.x == 0) {
        float den = clsn[b] * sqrtf(r2[0]);
        sim[blockIdx.x] = r1[0] / fmaxf(den, 1e-8f);
    }
}

// -------- top-K selection (descending, lowest-index tie-break) -------
__global__ __launch_bounds__(32)
void topk_kernel(const float* __restrict__ sim, int* __restrict__ idx)
{
    if (threadIdx.x != 0) return;
    int b = blockIdx.x;
    const float* s = sim + (long)b * NPATCH;
    float tmp[NPATCH];
    for (int i = 0; i < NPATCH; ++i) tmp[i] = s[i];
    for (int k = 0; k < KEEP; ++k) {
        int best = 0; float bv = tmp[0];
        for (int i = 1; i < NPATCH; ++i) if (tmp[i] > bv) { bv = tmp[i]; best = i; }
        idx[b * KEEP + k] = best;
        tmp[best] = -1e30f;
    }
}

// ---------------- gather kept tokens ---------------------------------
__global__ __launch_bounds__(256)
void gather_kernel(const float* __restrict__ x, const int* __restrict__ idx,
                   float* __restrict__ y, int Lold)
{
    int b = blockIdx.x / L1;
    int l = blockIdx.x % L1;
    const float* src = (l == 0)
        ? x + (long)b * Lold * DD
        : x + ((long)b * Lold + 1 + idx[b * KEEP + (l - 1)]) * DD;
    float* dst = y + ((long)b * L1 + l) * DD;
    for (int d = threadIdx.x; d < DD; d += 256) dst[d] = src[d];
}

// =====================================================================
// host side
// =====================================================================
static inline void launch_gemm(hipStream_t st,
    const float* A, int lda, long sAb, long sAh,
    const float* Bm, int ldb, long sBb, long sBh, int transB,
    float* C, int ldc, long sCb, long sCh,
    const float* bias,
    const float* res, int ldr, long sRb, long sRh,
    int M, int N, int K, int batch, int Hs, float alpha, int act)
{
    dim3 g((N + TN - 1) / TN, (M + TM - 1) / TM, batch);
    wmma_gemm_kernel<<<g, 256, 0, st>>>(A, lda, sAb, sAh, Bm, ldb, sBb, sBh, transB,
                                        C, ldc, sCb, sCh, bias, res, ldr, sRb, sRh,
                                        M, N, K, Hs, alpha, act);
}

extern "C" void kernel_launch(void* const* d_in, const int* in_sizes, int n_in,
                              void* d_out, int out_size, void* d_ws, size_t ws_size,
                              hipStream_t stream)
{
    const float* inputs  = (const float*)d_in[0];
    const float* patch_w = (const float*)d_in[1];
    const float* patch_b = (const float*)d_in[2];
    const float* cls_tok = (const float*)d_in[3];
    const float* pos_emb = (const float*)d_in[4];
    const float* qkv_w   = (const float*)d_in[5];
    const float* qkv_b   = (const float*)d_in[6];
    const float* proj_w  = (const float*)d_in[7];
    const float* proj_b  = (const float*)d_in[8];
    const float* ln1_g   = (const float*)d_in[9];
    const float* ln1_b   = (const float*)d_in[10];
    const float* ln2_g   = (const float*)d_in[11];
    const float* ln2_b   = (const float*)d_in[12];
    const float* mlp_w1  = (const float*)d_in[13];
    const float* mlp_b1  = (const float*)d_in[14];
    const float* mlp_w2  = (const float*)d_in[15];
    const float* mlp_b2  = (const float*)d_in[16];
    const float* norm_g  = (const float*)d_in[17];
    const float* norm_b  = (const float*)d_in[18];
    const float* head_w  = (const float*)d_in[19];
    const float* head_b  = (const float*)d_in[20];
    float* out = (float*)d_out;

    // workspace layout (floats)
    float* ws = (float*)d_ws;
    const size_t nx   = (size_t)BB * L0 * DD;             // token buffer
    const size_t nqkv = (size_t)BB * L0 * 3 * DD;         // qkv
    const size_t nsc  = (size_t)BB * HH * L0 * LP0;       // padded scores
    const size_t nmlp = (size_t)BB * L0 * DFF;            // mlp hidden

    float* x      = ws;
    float* x2     = x + nx;
    float* xn     = x2 + nx;        // LN output; reused as attention-O buffer
    float* qkv    = xn + nx;
    float* scores = qkv + nqkv;     // also patch-GEMM tmp
    float* mlph   = scores + nsc;   // also im2col tmp
    float* sim    = mlph + nmlp;
    float* clsn   = sim + (size_t)BB * NPATCH;
    int*   idxb   = (int*)(clsn + BB);

    const float scale = 0.125f;     // 1/sqrt(64)

    // ---- patch embedding ----
    {
        long tot = (long)BB * NPATCH * DD;
        im2col_kernel<<<(unsigned)((tot + 255) / 256), 256, 0, stream>>>(inputs, mlph);
        // (B*196, 768) @ (768, 768) + patch_b  -> scores tmp
        launch_gemm(stream, mlph, DD, 0, 0,
                    patch_w, DD, 0, 0, 0,
                    scores, DD, 0, 0,
                    patch_b, nullptr, 0, 0, 0,
                    BB * NPATCH, DD, DD, 1, 1, 1.0f, 0);
        build_x_kernel<<<BB * L0, 256, 0, stream>>>(scores, cls_tok, pos_emb, x);
    }

    float* xc = x;
    int L  = L0;
    int Lp = LP0;

    for (int n = 0; n < NLAY; ++n) {
        const int Mtok = BB * L;

        // LN1
        ln_kernel<<<Mtok, 256, 0, stream>>>(xc, ln1_g + (long)n * DD, ln1_b + (long)n * DD, xn, DD);

        // QKV: (B*L, 768) @ (768, 2304) + b
        launch_gemm(stream, xn, DD, 0, 0,
                    qkv_w + (long)n * DD * 3 * DD, 3 * DD, 0, 0, 0,
                    qkv, 3 * DD, 0, 0,
                    qkv_b + (long)n * 3 * DD, nullptr, 0, 0, 0,
                    Mtok, 3 * DD, DD, 1, 1, 1.0f, 0);

        // scores[b,h] = scale * Q K^T   (batched over z = b*H + h), ldc = Lp
        launch_gemm(stream,
                    qkv,              3 * DD, (long)L * 3 * DD, DHD,       // Q
                    qkv + DD,         3 * DD, (long)L * 3 * DD, DHD, 1,    // K (transposed)
                    scores,           Lp,     (long)HH * L * Lp, (long)L * Lp,
                    nullptr, nullptr, 0, 0, 0,
                    L, L, DHD, BB * HH, HH, scale, 0);

        softmax_kernel<<<BB * HH * L, 256, 0, stream>>>(scores, L, Lp);

        // O[b,h] = P V  -> packed into xn (B, L, 768) at column h*64
        launch_gemm(stream,
                    scores,           Lp,     (long)HH * L * Lp, (long)L * Lp,
                    qkv + 2 * DD,     3 * DD, (long)L * 3 * DD, DHD, 0,    // V
                    xn,               DD,     (long)L * DD, DHD,
                    nullptr, nullptr, 0, 0, 0,
                    L, DHD, L, BB * HH, HH, 1.0f, 0);

        // x = x + O @ proj_w + proj_b
        launch_gemm(stream, xn, DD, 0, 0,
                    proj_w + (long)n * DD * DD, DD, 0, 0, 0,
                    xc, DD, 0, 0,
                    proj_b + (long)n * DD,
                    xc, DD, 0, 0,
                    Mtok, DD, DD, 1, 1, 1.0f, 0);

        // LN2
        ln_kernel<<<Mtok, 256, 0, stream>>>(xc, ln2_g + (long)n * DD, ln2_b + (long)n * DD, xn, DD);

        // h = gelu(xn @ w1 + b1)
        launch_gemm(stream, xn, DD, 0, 0,
                    mlp_w1 + (long)n * DD * DFF, DFF, 0, 0, 0,
                    mlph, DFF, 0, 0,
                    mlp_b1 + (long)n * DFF, nullptr, 0, 0, 0,
                    Mtok, DFF, DD, 1, 1, 1.0f, 1);

        // x = x + h @ w2 + b2
        launch_gemm(stream, mlph, DFF, 0, 0,
                    mlp_w2 + (long)n * DFF * DD, DD, 0, 0, 0,
                    xc, DD, 0, 0,
                    mlp_b2 + (long)n * DD,
                    xc, DD, 0, 0,
                    Mtok, DD, DFF, 1, 1, 1.0f, 0);

        // ---- token pruning after layer SELL ----
        if (n == SELL) {
            clsnorm_kernel<<<BB, 256, 0, stream>>>(xc, clsn, L);
            sim_kernel<<<BB * (L - 1), 256, 0, stream>>>(xc, clsn, sim, L);
            topk_kernel<<<BB, 32, 0, stream>>>(sim, idxb);
            gather_kernel<<<BB * L1, 256, 0, stream>>>(xc, idxb, x2, L);
            xc = x2;
            L  = L1;
            Lp = LP1;
        }
    }

    // final LN + head
    ln_kernel<<<BB * L, 256, 0, stream>>>(xc, norm_g, norm_b, xn, DD);
    launch_gemm(stream, xn, DD, 0, 0,
                head_w, NCLS, 0, 0, 0,
                out, NCLS, 0, 0,
                head_b, nullptr, 0, 0, 0,
                BB * L, NCLS, DD, 1, 1, 1.0f, 0);
    (void)in_sizes; (void)n_in; (void)out_size; (void)ws_size;
}
